// Relation_Attention_52682068853339
// MI455X (gfx1250) — compile-verified, matrix operand order checked by reading, not run on hardware
//
#include <hip/hip_runtime.h>
#include <hip/hip_bf16.h>

#define BB 8
#define CC 64
#define C8 8
#define HH 128
#define WW 128
#define HWSZ (HH*WW)

typedef __attribute__((ext_vector_type(16))) __bf16 v16bf;
typedef __attribute__((ext_vector_type(8)))  float  v8f;
typedef __attribute__((ext_vector_type(4)))  unsigned int u32x4;
typedef __attribute__((ext_vector_type(8)))  int i32x8;
typedef __attribute__((ext_vector_type(4)))  int i32x4;

#if defined(__has_builtin)
#if __has_builtin(__builtin_amdgcn_tensor_load_to_lds) && __has_builtin(__builtin_amdgcn_s_wait_tensorcnt)
#define USE_TDM 1
#endif
#endif
#ifndef USE_TDM
#define USE_TDM 0
#endif

union FragU { unsigned int u[8]; v16bf v; };
union PackU { __bf16 b[8]; u32x4 u; };

// Pre-swizzle index for the B operand of v_wmma_f32_16x16x32_bf16 (wave32).
// Element (k in [0,32), column n): lane (n%16, hi=(k>>3)&1) reads VGPR j pair
// (k0,k0+1), k0 = (j&4?16:0)+hi*8+2*(j&3).  q = hi*8 + (k>=16)*4 + ((k>>1)&3).
__device__ __forceinline__ int bswz(int k, int n) {
  const int q = ((k >> 3) & 1) * 8 + ((k >> 4) & 1) * 4 + ((k >> 1) & 3);
  return n * 32 + q * 2 + (k & 1);
}

// A operand (16x32 bf16) from row-major LDS [M][ld]: two 16B dword runs per lane.
__device__ __forceinline__ v16bf frag_a_u(const __bf16* A, int ld, int mBase, int kBase, int lane) {
  const int idx16 = lane & 15;
  const int b0 = ((lane >> 4) & 1) * 4;
  const unsigned int* row = (const unsigned int*)(A + (size_t)(mBase + idx16) * ld + kBase);
  FragU f;
#pragma unroll
  for (int j = 0; j < 4; ++j) f.u[j] = row[b0 + j];
#pragma unroll
  for (int j = 0; j < 4; ++j) f.u[4 + j] = row[8 + b0 + j];
  return f.v;
}

// B operand from pre-swizzled LDS: 8 contiguous dwords per lane (2x ds_load_b128).
__device__ __forceinline__ v16bf frag_b_sw(const __bf16* Bsw, int nTot, int ks, int nBase, int lane) {
  const int idx16 = lane & 15;
  const int hi = (lane >> 4) & 1;
  const unsigned int* p =
      (const unsigned int*)(Bsw + (size_t)ks * nTot * 32 + (size_t)(nBase + idx16) * 32 + hi * 16);
  FragU f;
#pragma unroll
  for (int j = 0; j < 8; ++j) f.u[j] = p[j];
  return f.v;
}

// ---- Pass 1: q = Wq@Xq ; [k;v] = [Wk;Wv]@Xe  (1x1 convs as GEMM, K=64) ----
__global__ __launch_bounds__(128) void proj_qkv(
    const float* __restrict__ xq, const float* __restrict__ xe,
    const float* __restrict__ Wq, const float* __restrict__ bq,
    const float* __restrict__ Wk, const float* __restrict__ bk,
    const float* __restrict__ Wv, const float* __restrict__ bv,
    float* __restrict__ q_ws, float* __restrict__ k_ws, float* __restrict__ v_ws)
{
  __shared__ __bf16 We[80 * 64];        // rows 0..7 = Wk, 8..71 = Wv, 72..79 = 0 (A, row-major)
  __shared__ __bf16 Wqs[16 * 64];       // rows 0..7 = Wq, rest 0 (A, row-major)
  __shared__ __bf16 XeSw[2 * 64 * 32];  // B pre-swizzled, 2 K-chunks x 64 cols
  __shared__ __bf16 XqSw[2 * 64 * 32];

  const int tid = threadIdx.x;
  const int b = blockIdx.y;
  const int n0 = blockIdx.x * 64;

  for (int i = tid; i < 80 * 64; i += 128) {
    const int o = i >> 6, c = i & 63;
    float w = 0.f;
    if (o < 8) w = Wk[o * 64 + c];
    else if (o < 72) w = Wv[(o - 8) * 64 + c];
    We[i] = (__bf16)w;
  }
  for (int i = tid; i < 16 * 64; i += 128) {
    const int o = i >> 6, c = i & 63;
    Wqs[i] = (__bf16)(o < 8 ? Wq[o * 64 + c] : 0.f);
  }
  for (int i = tid; i < 64 * 64; i += 128) {
    const int c = i >> 6, n = i & 63;                       // n fastest -> coalesced global
    const int d = (c >> 5) * (64 * 32) + bswz(c & 31, n);
    XeSw[d] = (__bf16)xe[((size_t)b * CC + c) * HWSZ + n0 + n];
    XqSw[d] = (__bf16)xq[((size_t)b * CC + c) * HWSZ + n0 + n];
  }
  __syncthreads();

  const int wv = tid >> 5, lane = tid & 31;
  const int idx16 = lane & 15, hi = (lane >> 4) & 1;
  const int nBase = wv * 16;
  const int ng = n0 + nBase + idx16;

  { // q tile (rows 0..7 valid -> only hi==0 lanes store)
    v8f acc = {};
#pragma unroll
    for (int ks = 0; ks < 2; ++ks) {
      v16bf a  = frag_a_u(Wqs, 64, 0, ks * 32, lane);
      v16bf bm = frag_b_sw(XqSw, 64, ks, nBase, lane);
      acc = __builtin_amdgcn_wmma_f32_16x16x32_bf16(false, a, false, bm, (short)0, acc, false, false);
    }
    if (hi == 0) {
#pragma unroll
      for (int r = 0; r < 8; ++r)
        q_ws[((size_t)b * HWSZ + ng) * C8 + r] = acc[r] + bq[r];
    }
  }
#pragma unroll
  for (int mt = 0; mt < 5; ++mt) { // [k;v] tiles
    v8f acc = {};
#pragma unroll
    for (int ks = 0; ks < 2; ++ks) {
      v16bf a  = frag_a_u(We, 64, mt * 16, ks * 32, lane);
      v16bf bm = frag_b_sw(XeSw, 64, ks, nBase, lane);
      acc = __builtin_amdgcn_wmma_f32_16x16x32_bf16(false, a, false, bm, (short)0, acc, false, false);
    }
    if (mt == 0) {                 // rows 0..7 = k, rows 8..15 = v[0..7]
      if (hi == 0) {
#pragma unroll
        for (int r = 0; r < 8; ++r)
          k_ws[((size_t)b * HWSZ + ng) * C8 + r] = acc[r] + bk[r];
      } else {
#pragma unroll
        for (int r = 0; r < 8; ++r)
          v_ws[((size_t)b * HWSZ + ng) * CC + r] = acc[r] + bv[r];
      }
    } else if (mt < 4) {           // rows fully inside v
#pragma unroll
      for (int r = 0; r < 8; ++r) {
        const int vr = mt * 16 - 8 + r + hi * 8;
        v_ws[((size_t)b * HWSZ + ng) * CC + vr] = acc[r] + bv[vr];
      }
    } else if (hi == 0) {          // rows 64..71 = v[56..63]; 72..79 padding
#pragma unroll
      for (int r = 0; r < 8; ++r)
        v_ws[((size_t)b * HWSZ + ng) * CC + 56 + r] = acc[r] + bv[56 + r];
    }
  }
}

// ---- Pass 2: online softmax stats over concat(energy_H (diag masked), energy_W) ----
__global__ __launch_bounds__(128) void rowstats(
    const float* __restrict__ q_ws, const float* __restrict__ k_ws,
    float* __restrict__ mstat, float* __restrict__ rstat)
{
  __shared__ float Ksh[WW][C8];
  const int b = blockIdx.y, h = blockIdx.x, w = threadIdx.x;
  const size_t rowbase = ((size_t)b * HH + h) * WW;

  const float4 kr0 = *(const float4*)&k_ws[(rowbase + w) * C8];
  const float4 kr1 = *(const float4*)&k_ws[(rowbase + w) * C8 + 4];
  *(float4*)&Ksh[w][0] = kr0;
  *(float4*)&Ksh[w][4] = kr1;
  const float4 q0 = *(const float4*)&q_ws[(rowbase + w) * C8];
  const float4 q1 = *(const float4*)&q_ws[(rowbase + w) * C8 + 4];
  __syncthreads();

  float m = -__builtin_inff(), s = 0.f;
  for (int Hp = 0; Hp < HH; ++Hp) {           // column (H) part, diagonal masked
    if (Hp == h) continue;
    const float* kp = &k_ws[(((size_t)b * HH + Hp) * WW + w) * C8];
    const float4 k0 = *(const float4*)kp;
    const float4 k1 = *(const float4*)(kp + 4);
    float e = q0.x * k0.x + q0.y * k0.y + q0.z * k0.z + q0.w * k0.w
            + q1.x * k1.x + q1.y * k1.y + q1.z * k1.z + q1.w * k1.w;
    if (e > m) { s = s * __expf(m - e) + 1.f; m = e; }
    else       { s += __expf(e - m); }
  }
  for (int Wp = 0; Wp < WW; ++Wp) {           // row (W) part
    const float* kp = &Ksh[Wp][0];
    float e = q0.x * kp[0] + q0.y * kp[1] + q0.z * kp[2] + q0.w * kp[3]
            + q1.x * kp[4] + q1.y * kp[5] + q1.z * kp[6] + q1.w * kp[7];
    if (e > m) { s = s * __expf(m - e) + 1.f; m = e; }
    else       { s += __expf(e - m); }
  }
  mstat[rowbase + w] = m;
  rstat[rowbase + w] = 1.f / s;
}

// ---- Pass 3: per (b,w): E^T = K^T Q (WMMA, K=8 zero-padded) -> P -> out_H = V @ P^T ----
__global__ __launch_bounds__(256) void attnH(
    const float* __restrict__ q_ws, const float* __restrict__ k_ws,
    const float* __restrict__ v_ws, const float* __restrict__ mstat,
    const float* __restrict__ rstat, float* __restrict__ outH)
{
  __shared__ __align__(16) char smem[49 * 1024];
  __bf16* Kt  = (__bf16*)smem;               // 8KB  A: [H'=128][k=32] (cols 8..31 zero)
  __bf16* Qsw = (__bf16*)(smem + 8 * 1024);  // 8KB  B-swizzled: 1 K-chunk x 128 cols (h)
  __bf16* Vs  = (__bf16*)smem;               // 16KB A: [c=64][H'=128]  (aliases Kt+Qsw later)
  __bf16* Psw = (__bf16*)(smem + 16 * 1024); // 32KB B-swizzled: 4 K-chunks x 128 cols (h)
  float*  ms  = (float*)(smem + 48 * 1024);
  float*  rs  = (float*)(smem + 48 * 1024 + 512);

  const int tid = threadIdx.x;
  const int b = blockIdx.y, w = blockIdx.x;

  for (int i = tid; i < 128 * 32; i += 256) {
    const int x = i >> 5, c = i & 31;        // x: H' row of Kt AND h column of Qsw
    const size_t g = (((size_t)b * HH + x) * WW + w) * C8 + c;
    Kt[x * 32 + c]  = (__bf16)((c < C8) ? k_ws[g] : 0.f);
    Qsw[bswz(c, x)] = (__bf16)((c < C8) ? q_ws[g] : 0.f);
  }
  for (int i = tid; i < HH; i += 256) {
    ms[i] = mstat[((size_t)b * HH + i) * WW + w];
    rs[i] = rstat[((size_t)b * HH + i) * WW + w];
  }
  { // warm L2 for the V tile staged after the energy phase
    const int Hp = tid >> 1, c0 = (tid & 1) * 32;
    __builtin_prefetch(&v_ws[(((size_t)b * HH + Hp) * WW + w) * CC + c0], 0, 1);
  }
  __syncthreads();

  const int wv = tid >> 5, lane = tid & 31;
  const int idx16 = lane & 15, hi = (lane >> 4) & 1;

  // Energy (transposed): lane holds fixed h, 8 consecutive H' -> one b128 P store
#pragma unroll
  for (int t = 0; t < 8; ++t) {
    const int tile = wv + t * 8;
    const int mt = tile >> 3, nt = tile & 7;   // mt: H'-tile (k of PV), nt: h-tile
    v16bf a  = frag_a_u(Kt, 32, mt * 16, 0, lane);
    v16bf bm = frag_b_sw(Qsw, 128, 0, nt * 16, lane);
    v8f e = {};
    e = __builtin_amdgcn_wmma_f32_16x16x32_bf16(false, a, false, bm, (short)0, e, false, false);
    const int h = nt * 16 + idx16;             // fixed per lane
    const float mh = ms[h], rh = rs[h];
    const int K0 = mt * 16 + hi * 8;           // this lane's H' base (multiple of 8)
    PackU pk;
#pragma unroll
    for (int r = 0; r < 8; ++r) {
      const int Hp = K0 + r;
      const float p = (Hp == h) ? 0.f : __expf(e[r] - mh) * rh;
      pk.b[r] = (__bf16)p;
    }
    const int kk0 = K0 & 31;
    const int q0 = ((kk0 >> 3) & 1) * 8 + ((kk0 >> 4) & 1) * 4;
    *(u32x4*)&Psw[(K0 >> 5) * 4096 + h * 32 + q0 * 2] = pk.u;   // single ds_store_b128
  }
  __syncthreads();

  for (int i = tid; i < CC * HH; i += 256) {   // stage V (overwrites Kt/Qsw)
    const int Hp = i >> 6, c = i & 63;
    Vs[c * HH + Hp] = (__bf16)v_ws[(((size_t)b * HH + Hp) * WW + w) * CC + c];
  }
  __syncthreads();

  // out_H = V(64x128) @ P^T(128x128): 32 tiles x 4 K-steps
#pragma unroll
  for (int t = 0; t < 4; ++t) {
    const int tile = wv + t * 8;
    const int mt = tile >> 3, nt = tile & 7;   // mt: c-tile, nt: h-tile
    v8f acc = {};
#pragma unroll
    for (int ks = 0; ks < 4; ++ks) {
      v16bf a  = frag_a_u(Vs, HH, mt * 16, ks * 32, lane);
      v16bf bm = frag_b_sw(Psw, 128, ks, nt * 16, lane);
      acc = __builtin_amdgcn_wmma_f32_16x16x32_bf16(false, a, false, bm, (short)0, acc, false, false);
    }
    const int h = nt * 16 + idx16;
#pragma unroll
    for (int r = 0; r < 8; ++r) {
      const int c = mt * 16 + r + hi * 8;
      outH[(((size_t)b * HH + h) * WW + w) * CC + c] = acc[r];   // contiguous c-runs
    }
  }
}

// ---- Pass 4: per (b,h): out_W GEMM, TDM-load outH tile, LDS-transposed fused epilogue ----
__global__ __launch_bounds__(256) void attnW_combine(
    const float* __restrict__ q_ws, const float* __restrict__ k_ws,
    const float* __restrict__ v_ws, const float* __restrict__ mstat,
    const float* __restrict__ rstat, const float* __restrict__ outH,
    const float* __restrict__ xe, const float* __restrict__ xq,
    const float* __restrict__ g1p, const float* __restrict__ g2p,
    float* __restrict__ out)
{
  __shared__ __align__(16) char smem[49 * 1024];
  __bf16* Kt  = (__bf16*)smem;               // 8KB  A: [W'=128][k=32]
  __bf16* Qsw = (__bf16*)(smem + 8 * 1024);  // 8KB  B-swizzled (cols = w)
  __bf16* Vs  = (__bf16*)smem;               // 16KB A: [c=64][W'=128]
  __bf16* Psw = (__bf16*)(smem + 16 * 1024); // 32KB, later reused as outH tile / transpose buf
  float*  ms  = (float*)(smem + 48 * 1024);
  float*  rs  = (float*)(smem + 48 * 1024 + 512);

  const int tid = threadIdx.x;
  const int b = blockIdx.y, h = blockIdx.x;
  const size_t rowbase = ((size_t)b * HH + h) * WW;

  for (int i = tid; i < 128 * 32; i += 256) {
    const int x = i >> 5, c = i & 31;        // x: W' row of Kt AND w column of Qsw
    const size_t g = (rowbase + x) * C8 + c;
    Kt[x * 32 + c]  = (__bf16)((c < C8) ? k_ws[g] : 0.f);
    Qsw[bswz(c, x)] = (__bf16)((c < C8) ? q_ws[g] : 0.f);
  }
  for (int i = tid; i < WW; i += 256) {
    ms[i] = mstat[rowbase + i];
    rs[i] = rstat[rowbase + i];
  }
  {
    const int Wp = tid >> 1, c0 = (tid & 1) * 32;
    __builtin_prefetch(&v_ws[(rowbase + Wp) * CC + c0], 0, 1);
  }
  __syncthreads();

  const int wv = tid >> 5, lane = tid & 31;
  const int idx16 = lane & 15, hi = (lane >> 4) & 1;

#pragma unroll
  for (int t = 0; t < 8; ++t) {              // energy (no diagonal mask on W half)
    const int tile = wv + t * 8;
    const int mt = tile >> 3, nt = tile & 7;
    v16bf a  = frag_a_u(Kt, 32, mt * 16, 0, lane);
    v16bf bm = frag_b_sw(Qsw, 128, 0, nt * 16, lane);
    v8f e = {};
    e = __builtin_amdgcn_wmma_f32_16x16x32_bf16(false, a, false, bm, (short)0, e, false, false);
    const int w = nt * 16 + idx16;
    const float mw = ms[w], rw = rs[w];
    const int K0 = mt * 16 + hi * 8;
    PackU pk;
#pragma unroll
    for (int r = 0; r < 8; ++r) pk.b[r] = (__bf16)(__expf(e[r] - mw) * rw);
    const int kk0 = K0 & 31;
    const int q0 = ((kk0 >> 3) & 1) * 8 + ((kk0 >> 4) & 1) * 4;
    *(u32x4*)&Psw[(K0 >> 5) * 4096 + w * 32 + q0 * 2] = pk.u;
  }
  __syncthreads();

  for (int i = tid; i < CC * WW; i += 256) {
    const int Wp = i >> 6, c = i & 63;
    Vs[c * WW + Wp] = (__bf16)v_ws[(rowbase + Wp) * CC + c];
  }
  __syncthreads();

  v8f accs[4];
#pragma unroll
  for (int t = 0; t < 4; ++t) {              // out_W = V @ P^T, accumulators stay in regs
    const int tile = wv + t * 8;
    const int mt = tile >> 3, nt = tile & 7;
    v8f acc = {};
#pragma unroll
    for (int ks = 0; ks < 4; ++ks) {
      v16bf a  = frag_a_u(Vs, WW, mt * 16, ks * 32, lane);
      v16bf bm = frag_b_sw(Psw, 128, ks, nt * 16, lane);
      acc = __builtin_amdgcn_wmma_f32_16x16x32_bf16(false, a, false, bm, (short)0, acc, false, false);
    }
    accs[t] = acc;
  }
  __syncthreads();                            // all Psw reads done; region is reusable

  float* Th = (float*)(smem + 16 * 1024);     // outH tile [w][c] (linear 32KB block)
#if USE_TDM
  if (tid < 32) {                             // one wave programs the Tensor Data Mover
    const unsigned long long ga = (unsigned long long)(const void*)&outH[rowbase * CC];
    u32x4 g0;
    g0[0] = 1u;                               // count=1, user descriptor, no gather
    g0[1] = (unsigned)(size_t)(void*)Th;      // LDS byte address
    g0[2] = (unsigned)ga;                     // global_addr lo
    g0[3] = (unsigned)(ga >> 32) | (2u << 30);// global_addr hi | type=2 (image)
    i32x8 g1;
    g1[0] = (int)(2u << 16);                  // data_size = 4 bytes
    g1[1] = (int)(8192u << 16);               // tensor_dim0 = 8192 (lo16)
    g1[2] = (int)(1u << 16);                  // tensor_dim0 hi16=0 | tensor_dim1 = 1
    g1[3] = (int)(8192u << 16);               // tile_dim0 = 8192 (one linear row)
    g1[4] = 1;                                // tile_dim1 = 1, tile_dim2 = 0
    g1[5] = 8192;                             // tensor_dim0_stride (unused, 1 row)
    g1[6] = 0;
    g1[7] = 0;
    const i32x4 z4 = (i32x4)0;
#if __clang_major__ >= 23
    const i32x8 z8 = (i32x8)0;
    __builtin_amdgcn_tensor_load_to_lds(g0, g1, z4, z4, z8, 0);
#else
    __builtin_amdgcn_tensor_load_to_lds(g0, g1, z4, z4, 0);
#endif
    __builtin_amdgcn_s_wait_tensorcnt(0);
  }
  __syncthreads();
#endif

#pragma unroll
  for (int t = 0; t < 4; ++t) {              // add outH + 2.0 (out_LR + out_RL == 2.0 exactly)
    const int tile = wv + t * 8;
    const int mt = tile >> 3, nt = tile & 7;
    const int w = nt * 16 + idx16;
    const int cb = mt * 16 + hi * 8;
#if USE_TDM
    const float4 o0 = *(const float4*)&Th[w * 64 + cb];
    const float4 o1 = *(const float4*)&Th[w * 64 + cb + 4];
#else
    const float4 o0 = *(const float4*)&outH[(rowbase + w) * CC + cb];
    const float4 o1 = *(const float4*)&outH[(rowbase + w) * CC + cb + 4];
#endif
    accs[t][0] += o0.x + 2.0f; accs[t][1] += o0.y + 2.0f;
    accs[t][2] += o0.z + 2.0f; accs[t][3] += o0.w + 2.0f;
    accs[t][4] += o1.x + 2.0f; accs[t][5] += o1.y + 2.0f;
    accs[t][6] += o1.z + 2.0f; accs[t][7] += o1.w + 2.0f;
  }
  __syncthreads();

  float* Ts = (float*)(smem + 16 * 1024);    // transpose buffer [c][w]
#pragma unroll
  for (int t = 0; t < 4; ++t) {
    const int tile = wv + t * 8;
    const int mt = tile >> 3, nt = tile & 7;
    const int w = nt * 16 + idx16;
    const int cb = mt * 16 + hi * 8;
#pragma unroll
    for (int r = 0; r < 8; ++r) Ts[(cb + r) * WW + w] = accs[t][r];
  }
  __syncthreads();

  const float g1s = g1p[0], g2s = g2p[0];
  const size_t TOT = (size_t)BB * CC * HH * WW;
  for (int i = tid; i < CC * WW; i += 256) { // w fastest: contiguous x reads + out writes
    const int c = i >> 7, w = i & 127;
    const float o = Ts[c * WW + w];
    const size_t oidx = (((size_t)b * CC + c) * HH + h) * WW + w;
    out[oidx]       = g1s * o + xe[oidx];
    out[TOT + oidx] = g2s * o + xq[oidx];
  }
}

extern "C" void kernel_launch(void* const* d_in, const int* in_sizes, int n_in,
                              void* d_out, int out_size, void* d_ws, size_t ws_size,
                              hipStream_t stream) {
  const float* xe = (const float*)d_in[0];
  const float* xq = (const float*)d_in[1];
  const float* Wq = (const float*)d_in[2];
  const float* bq = (const float*)d_in[3];
  const float* Wk = (const float*)d_in[4];
  const float* bk = (const float*)d_in[5];
  const float* Wv = (const float*)d_in[6];
  const float* bv = (const float*)d_in[7];
  const float* g1 = (const float*)d_in[8];
  const float* g2 = (const float*)d_in[9];
  float* out = (float*)d_out;

  float* ws    = (float*)d_ws;
  float* q_ws  = ws;                                   // B*HW*8
  float* k_ws  = q_ws + (size_t)BB * HWSZ * C8;        // B*HW*8
  float* v_ws  = k_ws + (size_t)BB * HWSZ * C8;        // B*HW*64
  float* mstat = v_ws + (size_t)BB * HWSZ * CC;        // B*HW
  float* rstat = mstat + (size_t)BB * HWSZ;            // B*HW
  float* outH  = rstat + (size_t)BB * HWSZ;            // B*HW*64

  proj_qkv<<<dim3(HWSZ / 64, BB), 128, 0, stream>>>(xq, xe, Wq, bq, Wk, bk, Wv, bv,
                                                    q_ws, k_ws, v_ws);
  rowstats<<<dim3(HH, BB), WW, 0, stream>>>(q_ws, k_ws, mstat, rstat);
  attnH<<<dim3(WW, BB), 256, 0, stream>>>(q_ws, k_ws, v_ws, mstat, rstat, outH);
  attnW_combine<<<dim3(HH, BB), 256, 0, stream>>>(q_ws, k_ws, v_ws, mstat, rstat,
                                                  outH, xe, xq, g1, g2, out);
}
